// TopKSAE_18605798326360
// MI455X (gfx1250) — matrix operand order repaired and checked by reading, not compiled, not run on hardware
//
#include <hip/hip_runtime.h>

// ---------------------------------------------------------------------------
// TopK Sparse Autoencoder forward for MI455X (gfx1250, wave32, WMMA + TDM).
//
// Encoder GEMM x_norm[4096x1024] @ w_enc[1024x16384] = 137 GFLOP
//   -> v_wmma_f32_16x16x32_bf16 (bf16 in, fp32 accum)
//   -> tiles DMA'd into LDS by the Tensor Data Mover (tensor_load_to_lds,
//      TENSORcnt), double-buffered so DMA overlaps WMMA.
// Decoders are sparse row-gathers against w_dec (64 MB < 192 MB L2).
// ---------------------------------------------------------------------------

typedef __attribute__((ext_vector_type(16))) __bf16 v16bf;
typedef __attribute__((ext_vector_type(8)))  __bf16 v8bf;
typedef __attribute__((ext_vector_type(8)))  float  v8f;
typedef __attribute__((ext_vector_type(4)))  unsigned u32x4;
typedef __attribute__((ext_vector_type(8)))  int      i32x8;
typedef __attribute__((ext_vector_type(4)))  int      i32x4;

#define BATCH      4096
#define IN_DIM     1024
#define HID        16384
#define TOPK       64
#define KAUX       512
#define DEAD_THRESH_F 0.48828125f   // 2000/4096
#define LN_EPS     1e-5f

#if defined(__has_builtin)
#if __has_builtin(__builtin_amdgcn_tensor_load_to_lds) && \
    __has_builtin(__builtin_amdgcn_s_wait_tensorcnt)
#define HAVE_TDM 1
#endif
#endif

// ---------------- monotonic float<->uint ordering ----------------
__device__ __forceinline__ unsigned f2ord(float f) {
    unsigned b = __float_as_uint(f);
    return (b & 0x80000000u) ? ~b : (b | 0x80000000u);
}
__device__ __forceinline__ float ord2f(unsigned o) {
    unsigned b = (o & 0x80000000u) ? (o & 0x7FFFFFFFu) : ~o;
    return __uint_as_float(b);
}

// ---------------- 1) LayerNorm (ddof=1) + b_pre, emit bf16 ----------------
__global__ __launch_bounds__(256) void ln_kernel(
    const float* __restrict__ x, const float* __restrict__ b_pre,
    __bf16* __restrict__ xn, float* __restrict__ mu, float* __restrict__ stdv)
{
    __shared__ float red[256];
    const int r = blockIdx.x, tid = threadIdx.x;
    const float* row = x + (size_t)r * IN_DIM;
    float v[4];
    float s = 0.f;
    #pragma unroll
    for (int c = 0; c < 4; ++c) { v[c] = row[tid + c * 256]; s += v[c]; }
    red[tid] = s; __syncthreads();
    for (int st = 128; st > 0; st >>= 1) { if (tid < st) red[tid] += red[tid + st]; __syncthreads(); }
    const float mean = red[0] * (1.f / IN_DIM);
    __syncthreads();
    float sq = 0.f;
    #pragma unroll
    for (int c = 0; c < 4; ++c) { float d = v[c] - mean; sq += d * d; }
    red[tid] = sq; __syncthreads();
    for (int st = 128; st > 0; st >>= 1) { if (tid < st) red[tid] += red[tid + st]; __syncthreads(); }
    const float sd = sqrtf(red[0] * (1.f / (IN_DIM - 1)));
    if (tid == 0) { mu[r] = mean; stdv[r] = sd; }
    const float inv = 1.f / (sd + LN_EPS);
    #pragma unroll
    for (int c = 0; c < 4; ++c) {
        int col = tid + c * 256;
        xn[(size_t)r * IN_DIM + col] = (__bf16)((v[c] - mean) * inv - b_pre[col]);
    }
}

// ---------------- 2) w_enc [1024][16384] -> wencT bf16 [16384][1024] -------
__global__ __launch_bounds__(256) void wenc_transpose_kernel(
    const float* __restrict__ w_enc, __bf16* __restrict__ wt)
{
    __shared__ float tile[64][65];
    const int hb = blockIdx.x * 64;   // hidden
    const int kb = blockIdx.y * 64;   // input
    const int tid = threadIdx.x;
    for (int i = tid; i < 64 * 64; i += 256) {
        int kk = i >> 6, hh = i & 63;
        tile[kk][hh] = w_enc[(size_t)(kb + kk) * HID + hb + hh];   // coalesced in hh
    }
    __syncthreads();
    for (int i = tid; i < 64 * 64; i += 256) {
        int hh = i >> 6, kk = i & 63;
        wt[(size_t)(hb + hh) * IN_DIM + kb + kk] = (__bf16)tile[kk][hh]; // coalesced in kk
    }
}

// ---------------- 3) Encoder GEMM: pre = Xn @ WencT^T + b_enc --------------
// NT-GEMM: A [4096][1024] bf16 row-major, B(t) [16384][1024] bf16 row-major.
// Block tile 128x128x32, 8 waves in 2(M)x4(N), 8 WMMAs per wave per K-step.
#define BM 128
#define BN 128
#define BK 32
#define PAD 40   // bf16 per LDS row: 64B data + 16B pad => 80B, 16B-aligned chunks
#define KSTEPS (IN_DIM / BK)

#ifdef HAVE_TDM
// Tensor DMA: copy a 128x32 bf16 tile (row stride 1024 elems) into LDS with
// +16B padding after every 64B row (pad_interval code 3 = 16 DWORDs,
// pad_amount code 3 = 4 DWORDs)  => 80B LDS row stride == PAD.
__device__ __forceinline__ void tdm_load_tile(unsigned lds_addr,
                                              const __bf16* gptr,
                                              unsigned tensor_rows)
{
    const unsigned long long ga = (unsigned long long)gptr;
    u32x4 g0;
    g0[0] = 1u;                                    // count=1, user mode
    g0[1] = lds_addr;                              // LDS byte address
    g0[2] = (unsigned)(ga & 0xFFFFFFFFull);        // global_addr[31:0]
    g0[3] = (unsigned)((ga >> 32) & 0x01FFFFFFull) // global_addr[56:32]
          | 0x80000000u;                           // type=2 ("image")
    i32x8 g1;
    g1[0] = (int)((1u << 16)                       // data_size = 2 bytes
                | (1u << 20)                       // pad_enable
                | (3u << 22)                       // pad_interval: 16 DWORDs
                | (3u << 25));                     // pad_amount: 4 DWORDs
    g1[1] = (int)((IN_DIM & 0xFFFFu) << 16);       // tensor_dim0 lo16
    g1[2] = (int)(((unsigned)IN_DIM >> 16) | ((tensor_rows & 0xFFFFu) << 16));
    g1[3] = (int)((tensor_rows >> 16) | ((unsigned)BK << 16)); // tile_dim0=32
    g1[4] = BM;                                    // tile_dim1=128, tile_dim2=0
    g1[5] = IN_DIM;                                // tensor_dim0_stride lo32
    g1[6] = 0;
    g1[7] = 0;
    const i32x4 z4 = {0, 0, 0, 0};
#if __clang_major__ >= 23
    const i32x8 z8 = {0, 0, 0, 0, 0, 0, 0, 0};
    __builtin_amdgcn_tensor_load_to_lds(g0, g1, z4, z4, z8, 0);
#else
    __builtin_amdgcn_tensor_load_to_lds(g0, g1, z4, z4, 0);
#endif
}
#endif

// one K-step of fragments + 8 WMMAs for this wave
__device__ __forceinline__ void mma_step(const __bf16 (*tA)[PAD],
                                         const __bf16 (*tB)[PAD],
                                         int wm, int wn, int lr, int half,
                                         v8f acc[4][2])
{
    // A frag: lane<16 -> K 0..7 & 16..23 ; lane>=16 -> K 8..15 & 24..31
    v16bf afrag[4];
    #pragma unroll
    for (int mt = 0; mt < 4; ++mt) {
        int rowi = wm * 64 + mt * 16 + lr;
        int klo  = half * 8;
        union { v16bf v; v8bf h[2]; } u;
        u.h[0] = *(const v8bf*)&tA[rowi][klo];
        u.h[1] = *(const v8bf*)&tA[rowi][klo + 16];
        afrag[mt] = u.v;
    }
    // B frag: lane<16 -> K 0..7 & 16..23 ; lane>=16 -> K 8..15 & 24..31 (column = lr)
    v16bf bfrag[2];
    #pragma unroll
    for (int nt = 0; nt < 2; ++nt) {
        int rowi = wn * 32 + nt * 16 + lr;
        int kb   = half * 8;
        union { v16bf v; v8bf h[2]; } u;
        u.h[0] = *(const v8bf*)&tB[rowi][kb];
        u.h[1] = *(const v8bf*)&tB[rowi][kb + 16];
        bfrag[nt] = u.v;
    }
    #pragma unroll
    for (int mt = 0; mt < 4; ++mt)
        #pragma unroll
        for (int nt = 0; nt < 2; ++nt)
            acc[mt][nt] = __builtin_amdgcn_wmma_f32_16x16x32_bf16(
                false, afrag[mt], false, bfrag[nt],
                (short)0, acc[mt][nt], false, false);
}

__global__ __launch_bounds__(256) void enc_gemm_kernel(
    const __bf16* __restrict__ A, const __bf16* __restrict__ Bt,
    const float* __restrict__ b_enc, float* __restrict__ C)
{
    __shared__ __bf16 sA[2][BM][PAD];
    __shared__ __bf16 sB[2][BN][PAD];
    const int tid  = threadIdx.x;
    const int lane = tid & 31, w = tid >> 5;
    const int wm = w >> 2, wn = w & 3;       // 2 x 4 wave grid
    const int mBase = blockIdx.y * BM;       // batch
    const int nBase = blockIdx.x * BN;       // hidden
    const int half = lane >> 4, lr = lane & 15;

    v8f acc[4][2] = {};

    const __bf16* aTile = A  + (size_t)mBase * IN_DIM;
    const __bf16* bTile = Bt + (size_t)nBase * IN_DIM;

#ifdef HAVE_TDM
    // ---- TDM double-buffered pipeline: wave 0 drives the Tensor Data Mover.
    const unsigned ldsA[2] = {
        (unsigned)(unsigned long long)(void*)&sA[0][0][0],
        (unsigned)(unsigned long long)(void*)&sA[1][0][0] };
    const unsigned ldsB[2] = {
        (unsigned)(unsigned long long)(void*)&sB[0][0][0],
        (unsigned)(unsigned long long)(void*)&sB[1][0][0] };

    if (w == 0) {
        tdm_load_tile(ldsA[0], aTile, BATCH);
        tdm_load_tile(ldsB[0], bTile, HID);
        __builtin_amdgcn_s_wait_tensorcnt(0);
    }
    __syncthreads();

    for (int kk = 0; kk < KSTEPS; ++kk) {
        const int cur = kk & 1;
        if (w == 0 && kk + 1 < KSTEPS) {          // prefetch next tile via DMA
            const int k0n = (kk + 1) * BK;
            tdm_load_tile(ldsA[cur ^ 1], aTile + k0n, BATCH);
            tdm_load_tile(ldsB[cur ^ 1], bTile + k0n, HID);
        }
        mma_step(sA[cur], sB[cur], wm, wn, lr, half, acc);
        if (w == 0 && kk + 1 < KSTEPS)
            __builtin_amdgcn_s_wait_tensorcnt(0); // next buffer resident
        __syncthreads();
    }
#else
    // ---- fallback: straight-line synchronous staging (no divergent loops)
    const int rr0 = tid >> 2,          cc0 = (tid & 3) * 8;
    const int rr1 = (tid + 256) >> 2,  cc1 = ((tid + 256) & 3) * 8;
    for (int kk = 0; kk < KSTEPS; ++kk) {
        const int k0 = kk * BK;
        const uint4 a0 = *(const uint4*)(aTile + (size_t)rr0 * IN_DIM + k0 + cc0);
        const uint4 a1 = *(const uint4*)(aTile + (size_t)rr1 * IN_DIM + k0 + cc1);
        const uint4 b0 = *(const uint4*)(bTile + (size_t)rr0 * IN_DIM + k0 + cc0);
        const uint4 b1 = *(const uint4*)(bTile + (size_t)rr1 * IN_DIM + k0 + cc1);
        __syncthreads();                 // previous compute done before overwrite
        *(uint4*)&sA[0][rr0][cc0] = a0;
        *(uint4*)&sA[0][rr1][cc1] = a1;
        *(uint4*)&sB[0][rr0][cc0] = b0;
        *(uint4*)&sB[0][rr1][cc1] = b1;
        __syncthreads();
        mma_step(sA[0], sB[0], wm, wn, lr, half, acc);
    }
#endif

    // epilogue: C/D layout -> lane lr = column, VGPR r -> row r + 8*half
    #pragma unroll
    for (int nt = 0; nt < 2; ++nt) {
        int gcol = nBase + wn * 32 + nt * 16 + lr;
        float be = b_enc[gcol];
        #pragma unroll
        for (int mt = 0; mt < 4; ++mt) {
            #pragma unroll
            for (int rr = 0; rr < 8; ++rr) {
                int grow = mBase + wm * 64 + mt * 16 + rr + half * 8;
                C[(size_t)grow * HID + gcol] = acc[mt][nt][rr] + be;
            }
        }
    }
}

// ---------------- 4) utility: zero int buffer ------------------------------
__global__ void zero_i32_kernel(int* __restrict__ p, int n)
{
    int i = blockIdx.x * 256 + threadIdx.x;
    if (i < n) p[i] = 0;
}

// ---------------- 5) exact per-row top-K via radix descent ----------------
// One block per row. keys[] in dynamic LDS (H*4 B) + 256-entry scan buffer.
// Deterministic output ordering via prefix scans (no atomics).
__global__ __launch_bounds__(256) void topk_select_kernel(
    const float* __restrict__ pre, int H, int Kp,
    const int* __restrict__ dead,          // nullptr => main top-k
    int* __restrict__ out_idx, float* __restrict__ out_val,
    int* __restrict__ active)              // nullptr => aux top-k
{
    extern __shared__ unsigned smem[];
    unsigned* keys = smem;            // H
    unsigned* red  = smem + H;        // 256
    __shared__ int sG;

    const int tid = threadIdx.x;
    const int r   = blockIdx.x;
    const float* row = pre + (size_t)r * H;

    for (int i = tid; i < H; i += 256) {
        unsigned k;
        if (dead && dead[i] == 0) k = 0x007FFFFFu;   // ord(-inf)
        else                      k = f2ord(row[i]);
        keys[i] = k;
    }
    __syncthreads();

    // radix descent: largest T with count(keys >= T) >= Kp  (== Kp-th largest)
    unsigned prefix = 0u;
    for (int bit = 31; bit >= 0; --bit) {
        unsigned cand = prefix | (1u << bit);
        unsigned c = 0;
        for (int i = tid; i < H; i += 256) c += (keys[i] >= cand) ? 1u : 0u;
        red[tid] = c; __syncthreads();
        for (int s = 128; s > 0; s >>= 1) { if (tid < s) red[tid] += red[tid + s]; __syncthreads(); }
        if ((int)red[0] >= Kp) prefix = cand;
        __syncthreads();
    }
    const unsigned T = prefix;

    // thread-local counts of (> T) and (== T) over this thread's slice
    unsigned g = 0, e = 0;
    for (int i = tid; i < H; i += 256) {
        unsigned k = keys[i];
        g += (k > T) ? 1u : 0u;
        e += (k == T) ? 1u : 0u;
    }

    // inclusive scan of g (Hillis-Steele), deterministic
    red[tid] = g; __syncthreads();
    for (int s = 1; s < 256; s <<= 1) {
        unsigned v = (tid >= s) ? red[tid - s] : 0u;
        __syncthreads();
        red[tid] += v;
        __syncthreads();
    }
    unsigned gOff = red[tid] - g;
    if (tid == 255) sG = (int)red[255];
    __syncthreads();
    const int G = sG;
    __syncthreads();

    // inclusive scan of e
    red[tid] = e; __syncthreads();
    for (int s = 1; s < 256; s <<= 1) {
        unsigned v = (tid >= s) ? red[tid - s] : 0u;
        __syncthreads();
        red[tid] += v;
        __syncthreads();
    }
    unsigned eOff = red[tid] - e;

    int*   oi = out_idx + (size_t)r * Kp;
    float* ov = out_val + (size_t)r * Kp;
    unsigned gp = gOff, ep = eOff;
    for (int i = tid; i < H; i += 256) {
        unsigned k = keys[i];
        int pos = -1;
        if (k > T)       { pos = (int)gp++; }
        else if (k == T) { int p = (int)ep++; if (G + p < Kp) pos = G + p; }
        if (pos >= 0) {
            float v = ord2f(k);
            float a = v > 0.f ? v : 0.f;           // relu
            oi[pos] = i; ov[pos] = a;
            if (active && a > 0.f) active[i] = 1;  // benign multi-write of 1
        }
    }
}

// ---------------- 6) dead-neuron stats + num_dead --------------------------
__global__ __launch_bounds__(256) void dead_stats_kernel(
    const int* __restrict__ active, const int* __restrict__ stats_last,
    int* __restrict__ dead, float* __restrict__ num_dead_out)
{
    __shared__ int red[256];
    const int tid = threadIdx.x;
    int c = 0;
    for (int h = tid; h < HID; h += 256) {
        int du = active[h] ? 0 : 1;
        int st = stats_last[h] * du + 1;
        int dd = ((float)st > DEAD_THRESH_F) ? 1 : 0;
        dead[h] = dd;
        c += dd;
    }
    red[tid] = c; __syncthreads();
    for (int s = 128; s > 0; s >>= 1) { if (tid < s) red[tid] += red[tid + s]; __syncthreads(); }
    if (tid == 0) *num_dead_out = (float)red[0];
}

// ---------------- 7) sparse decode: out = (sum val*w_dec[idx] + b_pre)*std+mu
__global__ __launch_bounds__(256) void decode_kernel(
    const int* __restrict__ idx, const float* __restrict__ val, int Kp,
    const float* __restrict__ w_dec, const float* __restrict__ b_pre,
    const float* __restrict__ mu, const float* __restrict__ stdv,
    float* __restrict__ out)
{
    __shared__ int   sidx[KAUX];
    __shared__ float sval[KAUX];
    const int r = blockIdx.x, tid = threadIdx.x;
    for (int i = tid; i < Kp; i += 256) {
        sidx[i] = idx[(size_t)r * Kp + i];
        sval[i] = val[(size_t)r * Kp + i];
    }
    __syncthreads();

    float acc[4] = {0.f, 0.f, 0.f, 0.f};
    for (int j = 0; j < Kp; ++j) {
        float v = sval[j];                         // uniform across block
        const float* wrow = w_dec + (size_t)sidx[j] * IN_DIM;
        if (j + 1 < Kp)                            // L2-resident prefetch hint
            __builtin_prefetch(w_dec + (size_t)sidx[j + 1] * IN_DIM + tid, 0, 1);
        if (v != 0.f) {
            #pragma unroll
            for (int c = 0; c < 4; ++c) acc[c] += v * wrow[tid + c * 256];
        }
    }
    const float m = mu[r], s = stdv[r];
    #pragma unroll
    for (int c = 0; c < 4; ++c) {
        int col = tid + c * 256;
        out[(size_t)r * IN_DIM + col] = (acc[c] + b_pre[col]) * s + m;
    }
}

// ---------------------------------------------------------------------------
extern "C" void kernel_launch(void* const* d_in, const int* in_sizes, int n_in,
                              void* d_out, int out_size, void* d_ws, size_t ws_size,
                              hipStream_t stream)
{
    (void)in_sizes; (void)n_in; (void)out_size; (void)ws_size;

    const float* x      = (const float*)d_in[0];   // [4096,1024]
    const float* w_enc  = (const float*)d_in[1];   // [1024,16384]
    const float* w_dec  = (const float*)d_in[2];   // [16384,1024]
    const float* b_enc  = (const float*)d_in[3];   // [16384]
    const float* b_pre  = (const float*)d_in[4];   // [1024]
    const int*   stats  = (const int*)d_in[5];     // [16384]

    float* out        = (float*)d_out;
    float* out_recons = out;
    float* out_aux    = out + (size_t)BATCH * IN_DIM;
    float* out_ndead  = out + (size_t)2 * BATCH * IN_DIM;

    // workspace carve (aligned to 256 B)
    char* ws = (char*)d_ws;
    size_t off = 0;
    auto carve = [&](size_t bytes) -> void* {
        void* p = ws + off;
        off += (bytes + 255) & ~(size_t)255;
        return p;
    };
    float*  mu     = (float*)carve((size_t)BATCH * 4);
    float*  stdv   = (float*)carve((size_t)BATCH * 4);
    __bf16* xnbf   = (__bf16*)carve((size_t)BATCH * IN_DIM * 2);      // 8 MB
    __bf16* wencT  = (__bf16*)carve((size_t)HID * IN_DIM * 2);        // 32 MB
    float*  pre    = (float*)carve((size_t)BATCH * HID * 4);          // 256 MB
    int*    active = (int*)carve((size_t)HID * 4);
    int*    deadm  = (int*)carve((size_t)HID * 4);
    int*    tk_idx = (int*)carve((size_t)BATCH * TOPK * 4);
    float*  tk_val = (float*)carve((size_t)BATCH * TOPK * 4);
    int*    ax_idx = (int*)carve((size_t)BATCH * KAUX * 4);
    float*  ax_val = (float*)carve((size_t)BATCH * KAUX * 4);

    // 1) LayerNorm + bf16 conversion
    ln_kernel<<<BATCH, 256, 0, stream>>>(x, b_pre, xnbf, mu, stdv);

    // 2) w_enc -> transposed bf16
    wenc_transpose_kernel<<<dim3(HID / 64, IN_DIM / 64), 256, 0, stream>>>(w_enc, wencT);

    // 3) encoder GEMM with WMMA bf16 (+ TDM tile DMA)
    enc_gemm_kernel<<<dim3(HID / BN, BATCH / BM), 256, 0, stream>>>(xnbf, wencT, b_enc, pre);

    // 4) per-call zero of active flags (determinism)
    zero_i32_kernel<<<(HID + 255) / 256, 256, 0, stream>>>(active, HID);

    // 5) main top-K (K=64) + active marking
    const size_t shmem = (size_t)HID * 4 + 256 * 4;
    topk_select_kernel<<<BATCH, 256, shmem, stream>>>(
        pre, HID, TOPK, nullptr, tk_idx, tk_val, active);

    // 6) dead mask + num_dead
    dead_stats_kernel<<<1, 256, 0, stream>>>(active, stats, deadm, out_ndead);

    // 7) aux top-K (K=512) over dead-masked pre_acts
    topk_select_kernel<<<BATCH, 256, shmem, stream>>>(
        pre, HID, KAUX, deadm, ax_idx, ax_val, nullptr);

    // 8) decoders (sparse gathers, w_dec is L2-resident)
    decode_kernel<<<BATCH, 256, 0, stream>>>(tk_idx, tk_val, TOPK,
                                             w_dec, b_pre, mu, stdv, out_recons);
    decode_kernel<<<BATCH, 256, 0, stream>>>(ax_idx, ax_val, KAUX,
                                             w_dec, b_pre, mu, stdv, out_aux);
}
